// GSG_2808908611806
// MI455X (gfx1250) — matrix operand order repaired and checked by reading, not compiled, not run on hardware
//
#include <hip/hip_runtime.h>
#include <hip/hip_bf16.h>

#define LRELU_SLOPE 0.01f
#define BN_EPS 1e-5f

typedef __attribute__((ext_vector_type(16))) _Float16 v16h;
typedef __attribute__((ext_vector_type(8)))  _Float16 v8h;
typedef __attribute__((ext_vector_type(8)))  float    v8f;

__device__ __forceinline__ float lrelu_f(float x) { return x >= 0.f ? x : LRELU_SLOPE * x; }

// ---------------- depthwise 3x3, pad=1 ----------------
__global__ void dwconv3x3_k(const float* __restrict__ x, const float* __restrict__ w,
                            const float* __restrict__ b, float* __restrict__ y,
                            int C, int H, int W) {
  int p = blockIdx.x * blockDim.x + threadIdx.x;
  int HW = H * W;
  if (p >= HW) return;
  int c = blockIdx.y, n = blockIdx.z;
  int h = p / W, ww = p % W;
  const float* xs = x + ((long)n * C + c) * HW;
  const float* wc = w + c * 9;
  float acc = b ? b[c] : 0.f;
#pragma unroll
  for (int ky = 0; ky < 3; ky++) {
    int hh = h + ky - 1;
    if (hh < 0 || hh >= H) continue;
#pragma unroll
    for (int kx = 0; kx < 3; kx++) {
      int wx = ww + kx - 1;
      if (wx < 0 || wx >= W) continue;
      acc += xs[hh * W + wx] * wc[ky * 3 + kx];
    }
  }
  y[((long)n * C + c) * HW + p] = acc;
}

// ---------------- 1x1 (grouped) conv as WMMA GEMM ----------------
// Block = 8 waves sharing one (M-tile, image); each wave owns a 16-out-ch x 16-pixel tile.
// Weight tile (f16, zero-padded to Kp) + channel base-offset table staged in LDS once per
// block. Exact-K path (K % 32 == 0) has a select-free inner loop; padded path (K=8/16)
// zero-pads via cndmask selects. A-fragments load as two 16B LDS vectors.
__global__ __launch_bounds__(256)
void gemm1x1_wmma_k(const float* __restrict__ X, const float* __restrict__ Wt,
                    const float* __restrict__ bias,
                    const float* __restrict__ gamma, const float* __restrict__ beta,
                    const float* __restrict__ mean, const float* __restrict__ var,
                    float* __restrict__ Y,
                    int Cin, int HW, int K, int Gshuf, int coutTotal, int groups,
                    int bnlrelu, int concatMode, long sliceStride)
{
  __shared__ __align__(16) _Float16 wLds[16 * 128];  // 16 x Kp f16 weight tile
  __shared__ long long baseLds[128];                 // per-channel element base offset in X

  int coBase = blockIdx.y * 16;
  int n = blockIdx.z;
  int coutPerGroup = coutTotal / groups;
  int grp = coBase / coutPerGroup;
  int sBase = grp * K;
  int CoverG = Cin / Gshuf;
  int Kp = (K + 31) & ~31;

  // ---- cooperative staging (all 256 threads) ----
  for (int mm = 0; mm < 16; mm++)
    for (int l = threadIdx.x; l < Kp; l += 256) {
      int lc = (l < K) ? l : 0;
      float wv = Wt[(long)(coBase + mm) * K + lc];
      if (l >= K) wv = 0.f;
      wLds[mm * Kp + l] = (_Float16)wv;
    }
  for (int l = threadIdx.x; l < Kp; l += 256) {
    int ls = (l < K) ? l : 0;
    int s = sBase + ls;
    int c = (s % Gshuf) * CoverG + s / Gshuf;
    long b;
    if (!concatMode) b = ((long)n * Cin + c) * HW;
    else {
      int part = c >> 7;                       // concat order [u5,u4,u3,f2] = slices 4,3,2,1
      b = (long)(4 - part) * sliceStride + ((long)n * 128 + (c & 127)) * HW;
    }
    baseLds[l] = b;
  }
  __syncthreads();

  int lane = threadIdx.x & 31;
  int wave = threadIdx.x >> 5;
  int tile = blockIdx.x * 8 + wave;
  if (tile * 16 >= HW) return;                 // wave-uniform exit (after staging)
  int p0 = tile * 16;
  int m = lane & 15;                           // A row / B,D column index
  int half = lane >> 4;
  int p = p0 + m;

  v8f acc = {0.f, 0.f, 0.f, 0.f, 0.f, 0.f, 0.f, 0.f};

  if (K == Kp) {
    // ---- exact path: no zero-padding selects ----
    for (int kk = 0; kk < Kp; kk += 32) {
      v16h a, bm;
      v8h lo = *reinterpret_cast<const v8h*>(wLds + m * Kp + kk + half * 16);
      v8h hi = *reinterpret_cast<const v8h*>(wLds + m * Kp + kk + half * 16 + 8);
#pragma unroll
      for (int i = 0; i < 8; i++) { a[i] = lo[i]; a[8 + i] = hi[i]; }
#pragma unroll
      for (int v = 0; v < 8; v++) {
        int kb = kk + half * 16 + 2 * v;
        float x0 = X[baseLds[kb] + p];
        float x1 = X[baseLds[kb + 1] + p];
        bm[2 * v]     = (_Float16)x0;
        bm[2 * v + 1] = (_Float16)x1;
      }
      acc = __builtin_amdgcn_wmma_f32_16x16x32_f16(false, a, false, bm, (short)0, acc, false, false);
    }
  } else {
    // ---- padded path (K = 8 or 16 -> single 32-wide iteration) ----
    for (int kk = 0; kk < Kp; kk += 32) {
      v16h a, bm;
      v8h lo = *reinterpret_cast<const v8h*>(wLds + m * Kp + kk + half * 16);
      v8h hi = *reinterpret_cast<const v8h*>(wLds + m * Kp + kk + half * 16 + 8);
#pragma unroll
      for (int i = 0; i < 8; i++) { a[i] = lo[i]; a[8 + i] = hi[i]; }
#pragma unroll
      for (int v = 0; v < 8; v++) {
        int kb = kk + half * 16 + 2 * v;
        float x0 = X[baseLds[kb] + p];
        float x1 = X[baseLds[kb + 1] + p];
        if (kb >= K)     x0 = 0.f;
        if (kb + 1 >= K) x1 = 0.f;
        bm[2 * v]     = (_Float16)x0;
        bm[2 * v + 1] = (_Float16)x1;
      }
      acc = __builtin_amdgcn_wmma_f32_16x16x32_f16(false, a, false, bm, (short)0, acc, false, false);
    }
  }

#pragma unroll
  for (int j = 0; j < 8; j++) {
    int co = coBase + j + 8 * half;            // D layout: M = j + 8*half, N = lane%16
    float r = acc[j];
    if (bias) r += bias[co];
    if (bnlrelu) {
      float sc = gamma[co] * rsqrtf(var[co] + BN_EPS);
      r = (r - mean[co]) * sc + beta[co];
      r = lrelu_f(r);
    }
    Y[((long)n * coutTotal + co) * HW + p] = r;
  }
}

// ---------------- global average pool over HW ----------------
__global__ void gpool_k(const float* __restrict__ x, float* __restrict__ out, int C, int HW) {
  __shared__ float sm[256];
  int c = blockIdx.x, n = blockIdx.y;
  const float* xs = x + ((long)n * C + c) * HW;
  float s = 0.f;
  for (int i = threadIdx.x; i < HW; i += 256) s += xs[i];
  sm[threadIdx.x] = s;
  __syncthreads();
  for (int off = 128; off > 0; off >>= 1) {
    if (threadIdx.x < off) sm[threadIdx.x] += sm[threadIdx.x + off];
    __syncthreads();
  }
  if (threadIdx.x == 0) out[n * C + c] = sm[0] / (float)HW;
}

// ---------------- SE fully-connected chain ----------------
__global__ void se_fc_k(const float* __restrict__ pooled,
                        const float* __restrict__ w1, const float* __restrict__ b1,
                        const float* __restrict__ w2, const float* __restrict__ b2,
                        float* __restrict__ s, int C, int R) {
  __shared__ float hdn[32];
  int n = blockIdx.x, t = threadIdx.x;
  if (t < R) {
    float a = b1[t];
    for (int c = 0; c < C; c++) a += w1[t * C + c] * pooled[n * C + c];
    hdn[t] = lrelu_f(a);
  }
  __syncthreads();
  if (t < C) {
    float a = b2[t];
    for (int j = 0; j < R; j++) a += w2[t * R + j] * hdn[j];
    s[n * C + t] = 1.f / (1.f + expf(-a));
  }
}

// ---------------- x * s -> BN -> LeakyReLU ----------------
__global__ void scale_bn_lrelu_k(const float* __restrict__ x, const float* __restrict__ s,
                                 const float* __restrict__ gamma, const float* __restrict__ beta,
                                 const float* __restrict__ mean, const float* __restrict__ var,
                                 float* __restrict__ y, int C, int HW) {
  int p = blockIdx.x * blockDim.x + threadIdx.x;
  if (p >= HW) return;
  int c = blockIdx.y, n = blockIdx.z;
  long idx = ((long)n * C + c) * HW + p;
  float v = x[idx] * s[n * C + c];
  float sc = gamma[c] * rsqrtf(var[c] + BN_EPS);
  v = (v - mean[c]) * sc + beta[c];
  y[idx] = lrelu_f(v);
}

// ---------------- small dense 3x3 conv (CARAFE encoder) ----------------
__global__ void conv3x3_k(const float* __restrict__ x, const float* __restrict__ w,
                          const float* __restrict__ b, float* __restrict__ y,
                          int Cin, int Cout, int H, int W) {
  int p = blockIdx.x * blockDim.x + threadIdx.x;
  int HW = H * W;
  if (p >= HW) return;
  int co = blockIdx.y, n = blockIdx.z;
  int h = p / W, ww = p % W;
  float acc = b ? b[co] : 0.f;
  for (int ci = 0; ci < Cin; ci++) {
    const float* xs = x + ((long)n * Cin + ci) * HW;
    const float* wc = w + ((long)co * Cin + ci) * 9;
#pragma unroll
    for (int ky = 0; ky < 3; ky++) {
      int hh = h + ky - 1;
      if (hh < 0 || hh >= H) continue;
#pragma unroll
      for (int kx = 0; kx < 3; kx++) {
        int wx = ww + kx - 1;
        if (wx < 0 || wx >= W) continue;
        acc += xs[hh * W + wx] * wc[ky * 3 + kx];
      }
    }
  }
  y[((long)n * Cout + co) * HW + p] = acc;
}

// ---------------- CARAFE: softmax over k^2 + reassembly + pixel shuffle ----------------
// kt channel layout: c = kpos*4 + q ; softmax over kpos; out pixel (2h + q/2, 2w + q%2)
__global__ void carafe_up_k(const float* __restrict__ x, const float* __restrict__ kt,
                            float* __restrict__ out, int H, int W) {
  int p = blockIdx.x, n = blockIdx.y;
  int h = p / W, ww = p % W;
  int HW = H * W;
  __shared__ float wgt[9][4];
  int t = threadIdx.x;
  if (t < 4) {
    int q = t;
    float e[9], mx = -1e30f;
#pragma unroll
    for (int k = 0; k < 9; k++) {
      e[k] = kt[(long)(n * 36 + k * 4 + q) * HW + p];
      mx = fmaxf(mx, e[k]);
    }
    float sum = 0.f;
#pragma unroll
    for (int k = 0; k < 9; k++) { e[k] = expf(e[k] - mx); sum += e[k]; }
    float inv = 1.f / sum;
#pragma unroll
    for (int k = 0; k < 9; k++) wgt[k][q] = e[k] * inv;
  }
  __syncthreads();
  int c = t;                                     // 128 threads = channels
  const float* xs = x + ((long)n * 128 + c) * HW;
  float patch[9];
#pragma unroll
  for (int k = 0; k < 9; k++) {
    int hh = h + k / 3 - 1, wx = ww + k % 3 - 1;
    patch[k] = (hh >= 0 && hh < H && wx >= 0 && wx < W) ? xs[hh * W + wx] : 0.f;
  }
  int H2 = 2 * H, W2 = 2 * W;
  float* os = out + ((long)n * 128 + c) * H2 * W2;
#pragma unroll
  for (int q = 0; q < 4; q++) {
    float a = 0.f;
#pragma unroll
    for (int k = 0; k < 9; k++) a += patch[k] * wgt[k][q];
    os[(2 * h + q / 2) * W2 + (2 * ww + q % 2)] = a;
  }
}

// ================= host-side orchestration =================
struct CbnPtrs {
  const float *w, *b, *g, *bt, *mn, *vr;
  const float *w1, *b1, *w2, *b2;
  const float *sg, *sb, *sm, *sv;
};

static CbnPtrs cbn_at(void* const* d_in, int base) {
  // jax pytree (sorted-key) flatten order:
  // b, bn.beta, bn.gamma, bn.mean, bn.var, se.b1, se.b2,
  // se.bn.beta, se.bn.gamma, se.bn.mean, se.bn.var, se.w1, se.w2, w
  CbnPtrs P;
  P.b  = (const float*)d_in[base + 0];
  P.bt = (const float*)d_in[base + 1];
  P.g  = (const float*)d_in[base + 2];
  P.mn = (const float*)d_in[base + 3];
  P.vr = (const float*)d_in[base + 4];
  P.b1 = (const float*)d_in[base + 5];
  P.b2 = (const float*)d_in[base + 6];
  P.sb = (const float*)d_in[base + 7];
  P.sg = (const float*)d_in[base + 8];
  P.sm = (const float*)d_in[base + 9];
  P.sv = (const float*)d_in[base + 10];
  P.w1 = (const float*)d_in[base + 11];
  P.w2 = (const float*)d_in[base + 12];
  P.w  = (const float*)d_in[base + 13];
  return P;
}

static void launch_gemm(const float* X, const float* Wt, const float* bias,
                        const CbnPtrs* bn, float* Y, int Cin, int HW, int groups,
                        int Gshuf, int cout, int concatMode, long slice, hipStream_t st) {
  int K = Cin / groups;
  dim3 grid((HW / 16 + 7) / 8, cout / 16, 4);
  gemm1x1_wmma_k<<<grid, 256, 0, st>>>(X, Wt, bias,
      bn ? bn->g : nullptr, bn ? bn->bt : nullptr,
      bn ? bn->mn : nullptr, bn ? bn->vr : nullptr,
      Y, Cin, HW, K, Gshuf, cout, groups, bn ? 1 : 0, concatMode, slice);
}

static void run_cbn(const float* X, int Cin, int Gshuf, const CbnPtrs& P,
                    float* outBuf, float* big, float* pooled, float* svec,
                    int HW, int concatMode, long slice, hipStream_t st) {
  launch_gemm(X, P.w, P.b, &P, big, Cin, HW, 8, Gshuf, 128, concatMode, slice, st);
  gpool_k<<<dim3(128, 4), 256, 0, st>>>(big, pooled, 128, HW);
  se_fc_k<<<4, 128, 0, st>>>(pooled, P.w1, P.b1, P.w2, P.b2, svec, 128, 16);
  dim3 g2((HW + 255) / 256, 128, 4);
  scale_bn_lrelu_k<<<g2, 256, 0, st>>>(big, svec, P.sg, P.sb, P.sm, P.sv, outBuf, 128, HW);
}

static void run_carafe(const float* X, int H, int W,
                       const float* dw, const float* db, const float* ew, const float* eb,
                       const float* ow, const float* ob,
                       float* out, float* big, float* t1, float* t2, hipStream_t st) {
  int HW = H * W;
  launch_gemm(X, dw, db, nullptr, t1, 128, HW, 1, 128, 32, 0, 0, st);            // 128->32
  dim3 g((HW + 255) / 256, 36, 4);
  conv3x3_k<<<g, 256, 0, st>>>(t1, ew, eb, t2, 32, 36, H, W);                    // 32->36
  carafe_up_k<<<dim3(HW, 4), 128, 0, st>>>(X, t2, big, H, W);                    // reassembly
  launch_gemm(big, ow, ob, nullptr, out, 128, HW * 4, 1, 128, 128, 0, 0, st);    // 128->128
}

extern "C" void kernel_launch(void* const* d_in, const int* in_sizes, int n_in,
                              void* d_out, int out_size, void* d_ws, size_t ws_size,
                              hipStream_t stream) {
  (void)in_sizes; (void)n_in; (void)out_size; (void)ws_size;
  const float* x1 = (const float*)d_in[0];   // [4,64,192,192]
  const float* x2 = (const float*)d_in[1];   // [4,128,96,96]
  const float* x3 = (const float*)d_in[2];   // [4,256,48,48]
  const float* x4 = (const float*)d_in[3];   // [4,512,24,24]
  CbnPtrs cbn2 = cbn_at(d_in, 4);
  CbnPtrs cbn3 = cbn_at(d_in, 18);
  CbnPtrs cbn4 = cbn_at(d_in, 32);
  CbnPtrs cbn5 = cbn_at(d_in, 46);
  CbnPtrs cbnO = cbn_at(d_in, 60);
  const float* dw2_w  = (const float*)d_in[74];
  const float* dw3_w  = (const float*)d_in[75];
  const float* dw4_w  = (const float*)d_in[76];
  const float* dw5_b  = (const float*)d_in[77];
  const float* dw5_w  = (const float*)d_in[78];
  const float* down_b = (const float*)d_in[79];
  const float* down_w = (const float*)d_in[80];
  const float* enc_b  = (const float*)d_in[81];
  const float* enc_w  = (const float*)d_in[82];
  const float* out_b  = (const float*)d_in[83];
  const float* out_w  = (const float*)d_in[84];

  float* ws = (float*)d_ws;
  const long SLICE = 18874368L;              // 4*128*192*192
  float* big    = ws;                        // 18,874,368 (cbn pre-SE / carafe pre-out, time-shared)
  float* d2     = big    + 18874368L;        //  9,437,184
  float* d3     = d2     +  9437184L;        //  4,718,592
  float* d4     = d3     +  4718592L;        //  2,359,296
  float* d5     = d4     +  2359296L;        //  1,179,648
  float* f5     = d5     +  1179648L;        //    294,912
  float* f4     = f5     +   294912L;        //  1,179,648
  float* f3     = f4     +  1179648L;        //  4,718,592
  float* c1     = f3     +  4718592L;        //  1,179,648
  float* c2     = c1     +  1179648L;        //  4,718,592
  float* c3     = c2     +  4718592L;        //  4,718,592
  float* t1     = c3     +  4718592L;        //  1,179,648
  float* t2     = t1     +  1179648L;        //  1,327,104
  float* pooled = t2     +  1327104L;        //        512
  float* svec   = pooled +      512L;        //        512

  float* outF   = (float*)d_out;
  float* o_feat = outF;
  float* o_f2   = outF + SLICE;
  float* o_u3   = outF + 2 * SLICE;
  float* o_u4   = outF + 3 * SLICE;
  float* o_u5   = outF + 4 * SLICE;

  // depthwise 3x3 convs
  dwconv3x3_k<<<dim3((24 * 24 + 255) / 256, 512, 4), 256, 0, stream>>>(x4, dw5_w, dw5_b, d5, 512, 24, 24);
  dwconv3x3_k<<<dim3((48 * 48 + 255) / 256, 256, 4), 256, 0, stream>>>(x3, dw4_w, nullptr, d4, 256, 48, 48);
  dwconv3x3_k<<<dim3((96 * 96 + 255) / 256, 128, 4), 256, 0, stream>>>(x2, dw3_w, nullptr, d3, 128, 96, 96);
  dwconv3x3_k<<<dim3((192 * 192 + 255) / 256, 64, 4), 256, 0, stream>>>(x1, dw2_w, nullptr, d2, 64, 192, 192);

  // cbn blocks (Gshuf = gcd(cin,128))
  run_cbn(d5, 512, 128, cbn5, f5,  big, pooled, svec, 24 * 24,   0, 0, stream);
  run_cbn(d4, 256, 128, cbn4, f4,  big, pooled, svec, 48 * 48,   0, 0, stream);
  run_cbn(d3, 128, 128, cbn3, f3,  big, pooled, svec, 96 * 96,   0, 0, stream);
  run_cbn(d2,  64,  64, cbn2, o_f2, big, pooled, svec, 192 * 192, 0, 0, stream);

  // upsampling chains
  run_carafe(f3, 96, 96, down_w, down_b, enc_w, enc_b, out_w, out_b, o_u3, big, t1, t2, stream);
  run_carafe(f4, 48, 48, down_w, down_b, enc_w, enc_b, out_w, out_b, c3,   big, t1, t2, stream);
  run_carafe(c3, 96, 96, down_w, down_b, enc_w, enc_b, out_w, out_b, o_u4, big, t1, t2, stream);
  run_carafe(f5, 24, 24, down_w, down_b, enc_w, enc_b, out_w, out_b, c1,   big, t1, t2, stream);
  run_carafe(c1, 48, 48, down_w, down_b, enc_w, enc_b, out_w, out_b, c2,   big, t1, t2, stream);
  run_carafe(c2, 96, 96, down_w, down_b, enc_w, enc_b, out_w, out_b, o_u5, big, t1, t2, stream);

  // final cbn over concat([u5,u4,u3,f2]) read straight from the output slices
  run_cbn(outF, 512, 128, cbnO, o_feat, big, pooled, svec, 192 * 192, 1, SLICE, stream);
}
